// SelfAttention_37555194036870
// MI455X (gfx1250) — matrix operand order repaired and checked by reading, not compiled, not run on hardware
//
#include <hip/hip_runtime.h>
#include <math.h>

// ---------------------------------------------------------------------------
// Self-attention for MI455X (gfx1250), wave32 + v_wmma_f32_16x16x32_f16.
// B=2, S=2048, D=1024, H=16, DK=64.
// Pass 0: f32->f16 pre-conversion of activations + weights (elementwise).
// Pass 1: QKV projection GEMMs (pure f16 loads + WMMA, 32x64 tile/wave).
// Pass 2: flash attention: online softmax; row sums of P computed by an
//         extra WMMA against a ones-fragment (replaces 32 ds_bpermute/iter);
//         post-softmax multiplicative mask applied in A-layout so the
//         denominator stays mask-free; NT mask loads preserve L2.
// Pass 3: output projection GEMM -> f32 (NT stores).
// ---------------------------------------------------------------------------

#define S_LEN  2048
#define DMODEL 1024
#define HEADS  16
#define DKH    64
#define BATCH  2
#define MROWS  (BATCH * S_LEN)   // 4096

typedef _Float16 v16h __attribute__((ext_vector_type(16)));
typedef _Float16 v8h  __attribute__((ext_vector_type(8)));
typedef _Float16 v4h  __attribute__((ext_vector_type(4)));
typedef float    v8f  __attribute__((ext_vector_type(8)));
typedef float    v4f  __attribute__((ext_vector_type(4)));

__device__ __forceinline__ v8f wmma16(v16h a, v16h b, v8f c) {
  return __builtin_amdgcn_wmma_f32_16x16x32_f16(
      false, a, false, b, (short)0, c, false, false);
}

// ---------------------------------------------------------------------------
// Elementwise f32 -> f16 (n divisible by 4).
// ---------------------------------------------------------------------------
__global__ void __launch_bounds__(256)
cvt16_kernel(const float* __restrict__ src, _Float16* __restrict__ dst, int n4) {
  const int i = blockIdx.x * blockDim.x + threadIdx.x;
  if (i < n4) {
    v4f f = *(const v4f*)(src + (size_t)i * 4);
    v4h h;
#pragma unroll
    for (int e = 0; e < 4; ++e) h[e] = (_Float16)f[e];
    *(v4h*)(dst + (size_t)i * 4) = h;
  }
}

// ---------------------------------------------------------------------------
// Projection GEMM (f16 inputs): Y[m,n] = (sum_k X[m,k]*W[n,k] + bias[n])*scale
// One wave computes a 32x64 tile: 8 WMMAs per 32-wide K step.
// VMODE==0: store f16 row-major [4096,1024]   (Q, K)
// VMODE==1: store f16 per-head transposed Vt[b][h][dk][s_h]   (V)
// ---------------------------------------------------------------------------
template <int VMODE>
__global__ void __launch_bounds__(128)
proj16_kernel(const _Float16* __restrict__ X, const _Float16* __restrict__ W,
              const float* __restrict__ bias, _Float16* __restrict__ Y,
              float scale) {
  const int lane = threadIdx.x & 31;
  const int gw   = blockIdx.x * (blockDim.x >> 5) + (threadIdx.x >> 5); // 0..2047
  const int nt   = gw & 15;          // 16 n-tiles of 64
  const int mt   = gw >> 4;          // 128 m-tiles of 32
  const int lo   = lane & 15;
  const int hi   = lane >> 4;
  const int mbase = mt * 32;
  const int nbase = nt * 64;

  v8f c[2][4];
#pragma unroll
  for (int mi = 0; mi < 2; ++mi)
#pragma unroll
    for (int s = 0; s < 4; ++s)
#pragma unroll
      for (int r = 0; r < 8; ++r) c[mi][s][r] = 0.f;

  const _Float16* xr0 = X + (size_t)(mbase + lo) * DMODEL + hi * 16;
  const _Float16* xr1 = X + (size_t)(mbase + 16 + lo) * DMODEL + hi * 16;
  const _Float16* wr  = W + (size_t)(nbase + lo) * DMODEL + hi * 16;

  for (int k = 0; k < DMODEL; k += 32) {
    v8h a00 = *(const v8h*)(xr0 + k);
    v8h a01 = *(const v8h*)(xr0 + k + 8);
    v8h a10 = *(const v8h*)(xr1 + k);
    v8h a11 = *(const v8h*)(xr1 + k + 8);
    v16h bf[4];
#pragma unroll
    for (int s = 0; s < 4; ++s)
      bf[s] = *(const v16h*)(wr + (size_t)s * 16 * DMODEL + k);  // contiguous 32B

    v16h a[2];
#pragma unroll
    for (int e = 0; e < 8; ++e) {
      a[0][e] = a00[e]; a[0][8 + e] = a01[e];
      a[1][e] = a10[e]; a[1][8 + e] = a11[e];
    }
#pragma unroll
    for (int mi = 0; mi < 2; ++mi)
#pragma unroll
      for (int s = 0; s < 4; ++s)
        c[mi][s] = wmma16(a[mi], bf[s], c[mi][s]);
  }

  // ---- straight-line epilogue (affine addresses, no branches) -------------
#pragma unroll
  for (int mi = 0; mi < 2; ++mi)
#pragma unroll
    for (int s = 0; s < 4; ++s) {
      const int n = nbase + s * 16 + lo;
      const float bn = bias[n];
      if (VMODE == 0) {
        _Float16* yp = Y + (size_t)(mbase + mi * 16 + 8 * hi) * DMODEL + n;
#pragma unroll
        for (int r = 0; r < 8; ++r)
          yp[(size_t)r * DMODEL] = (_Float16)((c[mi][s][r] + bn) * scale);
      } else {
        // m = mbase + mi*16 + 8*hi + r ; all within one 128-row head block.
        const int m0   = mbase + mi * 16 + 8 * hi;
        const int bb   = m0 >> 11;
        const int srow = m0 & 2047;
        const int h    = srow >> 7;
        const int sh0  = ((srow & 127) << 4) + (n >> 6);  // +16 per r
        const int dk   = n & 63;
        _Float16* yp =
            Y + (((size_t)bb * HEADS + h) * DKH + dk) * S_LEN + sh0;
#pragma unroll
        for (int r = 0; r < 8; ++r)
          yp[r * 16] = (_Float16)((c[mi][s][r] + bn) * scale);
      }
    }
}

// ---------------------------------------------------------------------------
// Flash attention per (b, h, 16-row q-tile); one wave per q-tile.
// ---------------------------------------------------------------------------
__global__ void __launch_bounds__(64)
attn_kernel(const _Float16* __restrict__ Qp, const _Float16* __restrict__ Kp,
            const _Float16* __restrict__ Vt, const float* __restrict__ Ma_,
            const float* __restrict__ Mm_, _Float16* __restrict__ ctx) {
  __shared__ _Float16 plds[2][16][32];

  const int lane = threadIdx.x & 31;
  const int wv   = threadIdx.x >> 5;
  const int gw   = blockIdx.x * 2 + wv;   // 0..4095
  const int qt   = gw & 127;
  const int h    = (gw >> 7) & 15;
  const int b    = gw >> 11;
  const int lo   = lane & 15;
  const int hi   = lane >> 4;

  const _Float16* Qh = Qp + ((size_t)b * HEADS + h) * S_LEN * DKH;
  const _Float16* Kh = Kp + ((size_t)b * HEADS + h) * S_LEN * DKH;
  const _Float16* Vh = Vt + ((size_t)b * HEADS + h) * DKH * S_LEN;
  const float*    Ma = Ma_ + (size_t)b * S_LEN * S_LEN;
  const float*    Mm = Mm_ + (size_t)b * S_LEN * S_LEN;

  // Q A-fragments (dk 0..31, 32..63), kept resident; NORM pre-folded into Qp.
  v16h aq[2];
#pragma unroll
  for (int f = 0; f < 2; ++f) {
    const _Float16* p = Qh + (size_t)(qt * 16 + lo) * DKH + f * 32 + hi * 16;
    v8h c0 = *(const v8h*)p;
    v8h c1 = *(const v8h*)(p + 8);
#pragma unroll
    for (int e = 0; e < 8; ++e) { aq[f][e] = c0[e]; aq[f][8 + e] = c1[e]; }
  }

  // Constant ones B-fragment: row sums of P via WMMA (P @ ones).
  v16h ones;
#pragma unroll
  for (int e = 0; e < 16; ++e) ones[e] = (_Float16)1.0f;

  float mrun[8], lrun[8];
#pragma unroll
  for (int r = 0; r < 8; ++r) { mrun[r] = -INFINITY; lrun[r] = 0.f; }
  v8f acc[4];
#pragma unroll
  for (int d = 0; d < 4; ++d)
#pragma unroll
    for (int r = 0; r < 8; ++r) acc[d][r] = 0.f;

  for (int kb = 0; kb < S_LEN; kb += 32) {
    if (kb + 32 < S_LEN) {
      __builtin_prefetch(Kh + (size_t)(kb + 32 + lo) * DKH, 0, 1);
      __builtin_prefetch(Vh + (size_t)lo * S_LEN + kb + 32, 0, 1);
    }

    // ---- batched loads: K frags, V frags, additive mask (C-layout) --------
    v16h bk[2][2];
#pragma unroll
    for (int ntl = 0; ntl < 2; ++ntl)
#pragma unroll
      for (int kf = 0; kf < 2; ++kf)
        bk[ntl][kf] = *(const v16h*)(Kh + (size_t)(kb + ntl * 16 + lo) * DKH +
                                     kf * 32 + hi * 16);
    v16h bv[4];
#pragma unroll
    for (int dkt = 0; dkt < 4; ++dkt)
      bv[dkt] = *(const v16h*)(Vh + (size_t)(dkt * 16 + lo) * S_LEN + kb + hi * 16);

    float ma0[8], ma1[8];
#pragma unroll
    for (int r = 0; r < 8; ++r) {
      const size_t mo = (size_t)(qt * 16 + r + 8 * hi) * S_LEN + kb;
      ma0[r] = __builtin_nontemporal_load(Ma + mo + lo);
      ma1[r] = __builtin_nontemporal_load(Ma + mo + 16 + lo);
    }

    // Multiplicative mask directly in A-fragment layout (row=lo, key chunks).
    const float* mrow = Mm + (size_t)(qt * 16 + lo) * S_LEN + kb + hi * 16;
    v4f mg0 = __builtin_nontemporal_load((const v4f*)(mrow));
    v4f mg1 = __builtin_nontemporal_load((const v4f*)(mrow + 4));
    v4f mg2 = __builtin_nontemporal_load((const v4f*)(mrow + 8));
    v4f mg3 = __builtin_nontemporal_load((const v4f*)(mrow + 12));
    v16h mmA;
#pragma unroll
    for (int e = 0; e < 4; ++e) {
      mmA[e]      = (_Float16)mg0[e];
      mmA[4 + e]  = (_Float16)mg1[e];
      mmA[8 + e]  = (_Float16)mg2[e];
      mmA[12 + e] = (_Float16)mg3[e];
    }

    // ---- scores -----------------------------------------------------------
    v8f sc[2];
#pragma unroll
    for (int ntl = 0; ntl < 2; ++ntl) {
      v8f cz;
#pragma unroll
      for (int r = 0; r < 8; ++r) cz[r] = 0.f;
      cz = wmma16(aq[0], bk[ntl][0], cz);
      cz = wmma16(aq[1], bk[ntl][1], cz);
      sc[ntl] = cz;
    }

    // ---- online softmax max + raw p (row = r + 8*hi, key col = lo) --------
    float rescl[8];
#pragma unroll
    for (int r = 0; r < 8; ++r) {
      float s0 = sc[0][r] + ma0[r];
      float s1 = sc[1][r] + ma1[r];
      float mx = fmaxf(s0, s1);
#pragma unroll
      for (int d = 1; d < 16; d <<= 1) mx = fmaxf(mx, __shfl_xor(mx, d, 32));
      const float mnew = fmaxf(mrun[r], mx);
      rescl[r] = __expf(mrun[r] - mnew);            // 0 on first tile
      mrun[r]  = mnew;
      const float p0 = __expf(s0 - mnew);
      const float p1 = __expf(s1 - mnew);
#pragma unroll
      for (int d = 0; d < 4; ++d) acc[d][r] *= rescl[r];
      plds[wv][r + 8 * hi][lo]      = (_Float16)p0;   // raw p (no mask1)
      plds[wv][r + 8 * hi][16 + lo] = (_Float16)p1;
    }
    asm volatile("s_wait_dscnt 0" ::: "memory");

    // ---- raw P (C-layout in LDS) -> A-fragment ----------------------------
    v16h ap_raw;
    {
      const _Float16* pl = &plds[wv][lo][hi * 16];
      v8h c0 = *(const v8h*)pl;
      v8h c1 = *(const v8h*)(pl + 8);
#pragma unroll
      for (int e = 0; e < 8; ++e) { ap_raw[e] = c0[e]; ap_raw[8 + e] = c1[e]; }
    }

    // ---- row sums of raw P via WMMA against ones (denominator) ------------
    v8f lz;
#pragma unroll
    for (int r = 0; r < 8; ++r) lz[r] = 0.f;
    lz = wmma16(ap_raw, ones, lz);
#pragma unroll
    for (int r = 0; r < 8; ++r) lrun[r] = lrun[r] * rescl[r] + lz[r];

    // ---- apply post-softmax multiplicative mask in A-layout ---------------
    v16h ap = ap_raw * mmA;   // packed f16 muls

    // ---- acc += (P*mask1) @ V ---------------------------------------------
#pragma unroll
    for (int dkt = 0; dkt < 4; ++dkt)
      acc[dkt] = wmma16(ap, bv[dkt], acc[dkt]);
  }

  // ---- epilogue -----------------------------------------------------------
  _Float16* ch = ctx + ((size_t)b * HEADS + h) * S_LEN * DKH;
#pragma unroll
  for (int dkt = 0; dkt < 4; ++dkt)
#pragma unroll
    for (int r = 0; r < 8; ++r) {
      const float o = acc[dkt][r] / lrun[r];
      ch[(size_t)(qt * 16 + r + 8 * hi) * DKH + dkt * 16 + lo] = (_Float16)o;
    }
}

// ---------------------------------------------------------------------------
// Output projection (f16 x f16 -> f32): out[m,n] = sum_k ctx[m,k]*Wo[n,k]+bo[n]
// 32x64 tile per wave; final result stored non-temporally (never re-read).
// ---------------------------------------------------------------------------
__global__ void __launch_bounds__(128)
outproj16_kernel(const _Float16* __restrict__ Xh, const _Float16* __restrict__ W,
                 const float* __restrict__ bias, float* __restrict__ Y) {
  const int lane = threadIdx.x & 31;
  const int gw   = blockIdx.x * (blockDim.x >> 5) + (threadIdx.x >> 5);
  const int nt   = gw & 15;
  const int mt   = gw >> 4;
  const int lo   = lane & 15;
  const int hi   = lane >> 4;
  const int mbase = mt * 32;
  const int nbase = nt * 64;

  v8f c[2][4];
#pragma unroll
  for (int mi = 0; mi < 2; ++mi)
#pragma unroll
    for (int s = 0; s < 4; ++s)
#pragma unroll
      for (int r = 0; r < 8; ++r) c[mi][s][r] = 0.f;

  const _Float16* xr0 = Xh + (size_t)(mbase + lo) * DMODEL + hi * 16;
  const _Float16* xr1 = Xh + (size_t)(mbase + 16 + lo) * DMODEL + hi * 16;
  const _Float16* wr  = W + (size_t)(nbase + lo) * DMODEL + hi * 16;

  for (int k = 0; k < DMODEL; k += 32) {
    v8h a00 = *(const v8h*)(xr0 + k);
    v8h a01 = *(const v8h*)(xr0 + k + 8);
    v8h a10 = *(const v8h*)(xr1 + k);
    v8h a11 = *(const v8h*)(xr1 + k + 8);
    v16h bf[4];
#pragma unroll
    for (int s = 0; s < 4; ++s)
      bf[s] = *(const v16h*)(wr + (size_t)s * 16 * DMODEL + k);

    v16h a[2];
#pragma unroll
    for (int e = 0; e < 8; ++e) {
      a[0][e] = a00[e]; a[0][8 + e] = a01[e];
      a[1][e] = a10[e]; a[1][8 + e] = a11[e];
    }
#pragma unroll
    for (int mi = 0; mi < 2; ++mi)
#pragma unroll
      for (int s = 0; s < 4; ++s)
        c[mi][s] = wmma16(a[mi], bf[s], c[mi][s]);
  }

#pragma unroll
  for (int mi = 0; mi < 2; ++mi)
#pragma unroll
    for (int s = 0; s < 4; ++s) {
      const int n = nbase + s * 16 + lo;
      const float bn = bias[n];
      float* yp = Y + (size_t)(mbase + mi * 16 + 8 * hi) * DMODEL + n;
#pragma unroll
      for (int r = 0; r < 8; ++r)
        __builtin_nontemporal_store(c[mi][s][r] + bn, yp + (size_t)r * DMODEL);
    }
}

// ---------------------------------------------------------------------------
extern "C" void kernel_launch(void* const* d_in, const int* in_sizes, int n_in,
                              void* d_out, int out_size, void* d_ws,
                              size_t ws_size, hipStream_t stream) {
  const float* q    = (const float*)d_in[0];
  const float* k    = (const float*)d_in[1];
  const float* v    = (const float*)d_in[2];
  const float* ma   = (const float*)d_in[3];
  const float* mm   = (const float*)d_in[4];
  const float* wq_w = (const float*)d_in[5];
  const float* wq_b = (const float*)d_in[6];
  const float* wk_w = (const float*)d_in[7];
  const float* wk_b = (const float*)d_in[8];
  const float* wv_w = (const float*)d_in[9];
  const float* wv_b = (const float*)d_in[10];
  const float* wo_w = (const float*)d_in[11];
  const float* wo_b = (const float*)d_in[12];
  float* out = (float*)d_out;

  const size_t ACT = (size_t)MROWS * DMODEL;        // 4M elements
  const size_t WEL = (size_t)DMODEL * DMODEL;       // 1M elements
  _Float16* base = (_Float16*)d_ws;
  _Float16* Xq16 = base;                // 4M
  _Float16* Xk16 = Xq16 + ACT;          // 4M
  _Float16* Xv16 = Xk16 + ACT;          // 4M
  _Float16* Wq16 = Xv16 + ACT;          // 1M
  _Float16* Wk16 = Wq16 + WEL;
  _Float16* Wv16 = Wk16 + WEL;
  _Float16* Wo16 = Wv16 + WEL;
  _Float16* Qp   = Wo16 + WEL;          // 4M
  _Float16* Kp   = Qp + ACT;
  _Float16* Vt   = Kp + ACT;
  _Float16* ctx  = Vt + ACT;

  // ---- pass 0: f32 -> f16 pre-conversion ----------------------------------
  const int actn4 = (int)(ACT / 4), weln4 = (int)(WEL / 4);
  cvt16_kernel<<<(actn4 + 255) / 256, 256, 0, stream>>>(q, Xq16, actn4);
  cvt16_kernel<<<(actn4 + 255) / 256, 256, 0, stream>>>(k, Xk16, actn4);
  cvt16_kernel<<<(actn4 + 255) / 256, 256, 0, stream>>>(v, Xv16, actn4);
  cvt16_kernel<<<(weln4 + 255) / 256, 256, 0, stream>>>(wq_w, Wq16, weln4);
  cvt16_kernel<<<(weln4 + 255) / 256, 256, 0, stream>>>(wk_w, Wk16, weln4);
  cvt16_kernel<<<(weln4 + 255) / 256, 256, 0, stream>>>(wv_w, Wv16, weln4);
  cvt16_kernel<<<(weln4 + 255) / 256, 256, 0, stream>>>(wo_w, Wo16, weln4);

  // ---- pass 1: projections (NORM=0.125 folded into Q) ---------------------
  const int waves = (MROWS / 32) * (DMODEL / 64);   // 2048
  dim3 gblk(128), ggrd(waves / 4);
  proj16_kernel<0><<<ggrd, gblk, 0, stream>>>(Xq16, Wq16, wq_b, Qp, 0.125f);
  proj16_kernel<0><<<ggrd, gblk, 0, stream>>>(Xk16, Wk16, wk_b, Kp, 1.0f);
  proj16_kernel<1><<<ggrd, gblk, 0, stream>>>(Xv16, Wv16, wv_b, Vt, 1.0f);

  // ---- pass 2: flash attention --------------------------------------------
  const int awaves = BATCH * HEADS * (S_LEN / 16);  // 4096
  attn_kernel<<<awaves / 2, 64, 0, stream>>>(Qp, Kp, Vt, ma, mm, ctx);

  // ---- pass 3: output projection ------------------------------------------
  outproj16_kernel<<<ggrd, gblk, 0, stream>>>(ctx, Wo16, wo_b, out);
}
